// GraphConvolution_7430293422185
// MI455X (gfx1250) — matrix operand order repaired and checked by reading, not compiled, not run on hardware
//
#include <hip/hip_runtime.h>
#include <hip/hip_bf16.h>

// ---- CDNA5 WMMA vector types --------------------------------------------
typedef __attribute__((ext_vector_type(16))) __bf16 v16bf;
typedef __attribute__((ext_vector_type(8)))  float  v8f;

#define V   2048     // nodes
#define D   64       // feature dim (in == out)
#define NB  64       // A*B = 8*8 batches
#define ADJT_OFF ((size_t)8192)                       // after r_inv
#define TT_OFF   ((size_t)8192 + (size_t)V * V * 2)   // after adjT (bf16)

// =========================================================================
// K1: r_inv[j] = rsqrt( sum_i adj[i][j] )   (column sums)
// grid(64), block(32,8)
// =========================================================================
__global__ void colsum_rinv_kernel(const float* __restrict__ adj,
                                   float* __restrict__ rinv) {
    __shared__ float part[8][32];
    const int tx = threadIdx.x;          // 0..31  -> column within block
    const int ty = threadIdx.y;          // 0..7   -> row strip
    const int j  = blockIdx.x * 32 + tx;
    float s = 0.f;
    for (int i = ty; i < V; i += 8)
        s += adj[(size_t)i * V + j];     // coalesced across tx
    part[ty][tx] = s;
    __syncthreads();
    if (ty == 0) {
        float t = 0.f;
#pragma unroll
        for (int r = 0; r < 8; ++r) t += part[r][tx];
        rinv[j] = rsqrtf(t);
    }
}

// =========================================================================
// K2: adjT[j][v] = bf16( adj[v][j] * r_inv[v] )   (WMMA-A operand, row-major)
// LDS-tiled 32x32 transpose. grid(64,64), block(32,8)
// =========================================================================
__global__ void transpose_scale_kernel(const float* __restrict__ adj,
                                       const float* __restrict__ rinv,
                                       unsigned short* __restrict__ adjT_u16) {
    __bf16* adjT = (__bf16*)adjT_u16;
    __shared__ float tile[32 * 33];      // +1 pad: conflict-free transpose
    const int tx = threadIdx.x, ty = threadIdx.y;
    const int vbase = blockIdx.x * 32;
    const int jbase = blockIdx.y * 32;
#pragma unroll
    for (int r = 0; r < 4; ++r) {
        const int lv = ty + r * 8;
        const int v  = vbase + lv;
        tile[lv * 33 + tx] = adj[(size_t)v * V + jbase + tx] * rinv[v];
    }
    __syncthreads();
#pragma unroll
    for (int r = 0; r < 4; ++r) {
        const int lj = ty + r * 8;
        const int j  = jbase + lj;
        adjT[(size_t)j * V + vbase + tx] = (__bf16)tile[tx * 33 + lj];
    }
}

// =========================================================================
// K3: tT[b][d][v] = bf16( sum_k x[b][v][k] * w[k][d] )   (WMMA-B operand,
// stored K-contiguous per output column). grid(64 vchunks, 64 batches),
// block(256).
// =========================================================================
__global__ void project_kernel(const float* __restrict__ x,
                               const float* __restrict__ w,
                               unsigned short* __restrict__ tT_u16) {
    __bf16* tT = (__bf16*)tT_u16;
    __shared__ float xs[32 * 65];        // 32 node rows, padded
    __shared__ float wsm[64 * 64];
    const int tid   = threadIdx.x;       // 256
    const int b     = blockIdx.y;
    const int vbase = blockIdx.x * 32;
    const float* xb = x + (size_t)b * V * D + (size_t)vbase * D;
#pragma unroll
    for (int t = 0; t < 8; ++t) {        // 32*64 floats of x
        const int idx = tid + t * 256;
        xs[(idx >> 6) * 65 + (idx & 63)] = xb[idx];
    }
#pragma unroll
    for (int t = 0; t < 16; ++t)         // full 64x64 weight
        wsm[tid + t * 256] = w[tid + t * 256];
    __syncthreads();

    const int lane = tid & 31;           // node v within chunk
    const int dg   = tid >> 5;           // 8 output-feature octets
    float acc[8];
#pragma unroll
    for (int i = 0; i < 8; ++i) acc[i] = 0.f;
    for (int k = 0; k < 64; ++k) {
        const float xv = xs[lane * 65 + k];     // conflict-free (stride 65)
#pragma unroll
        for (int i = 0; i < 8; ++i)             // wsm read is broadcast
            acc[i] += xv * wsm[k * 64 + dg * 8 + i];
    }
#pragma unroll
    for (int i = 0; i < 8; ++i) {
        const int d = dg * 8 + i;
        tT[(size_t)b * (V * D) + (size_t)d * V + vbase + lane] = (__bf16)acc[i];
    }
}

// =========================================================================
// K4: out[b][j][d] = relu( rinv[j] * sum_v adjT[j][v]*tT[b][d][v] + bias[d] )
// One wave per 64x64 output tile: 4x4 accumulator tiles, K-loop step 32,
// 16 x v_wmma_f32_16x16x32_bf16 per step. grid(256), block(256) = 8 waves.
// =========================================================================
__global__ void __launch_bounds__(256)
gemm_agg_kernel(const unsigned short* __restrict__ adjT_u16,
                const unsigned short* __restrict__ tT_u16,
                const float* __restrict__ rinv,
                const float* __restrict__ bias,
                float* __restrict__ out) {
    const __bf16* adjT = (const __bf16*)adjT_u16;
    const __bf16* tT   = (const __bf16*)tT_u16;

    const int tid   = threadIdx.x;
    const int lane  = tid & 31;
    const int wid   = blockIdx.x * 8 + (tid >> 5);  // 0..2047 wave tiles
    const int batch = wid & 63;                     // 64 batches
    const int m     = (wid >> 6) * 64;              // 32 row tiles of 64
    const int lh    = lane & 15;
    const int hi    = lane >> 4;                    // lane half select

    // A operand (16x32 bf16): lane row M = lh; lanes<16 take K 0..15,
    // lanes>=16 take K 16..31  -> two contiguous b128 loads.
    const __bf16* pA[4];
#pragma unroll
    for (int s = 0; s < 4; ++s)
        pA[s] = adjT + (size_t)(m + s * 16 + lh) * V + (hi ? 16 : 0);

    // B operand (32x16 bf16): lane col N = lh; lanes<16 K=0..15,
    // lanes>=16 K=16..31 -> one contiguous 32B load (two b128).
    const __bf16* tb = tT + (size_t)batch * (V * D);
    const __bf16* pB[4];
#pragma unroll
    for (int t = 0; t < 4; ++t)
        pB[t] = tb + (size_t)(t * 16 + lh) * V + (hi ? 16 : 0);

    v8f acc[4][4];
#pragma unroll
    for (int s = 0; s < 4; ++s)
#pragma unroll
        for (int t = 0; t < 4; ++t) acc[s][t] = (v8f)(0.f);

    for (int kb = 0; kb < V; kb += 32) {
        v16bf bfr[4];
#pragma unroll
        for (int t = 0; t < 4; ++t) {
            const uint4* p = (const uint4*)(pB[t] + kb);
            ((uint4*)&bfr[t])[0] = p[0];
            ((uint4*)&bfr[t])[1] = p[1];
        }
#pragma unroll
        for (int s = 0; s < 4; ++s) {
            v16bf afr;
            ((uint4*)&afr)[0] = *(const uint4*)(pA[s] + kb);
            ((uint4*)&afr)[1] = *(const uint4*)(pA[s] + kb + 8);
#pragma unroll
            for (int t = 0; t < 4; ++t)
                acc[s][t] = __builtin_amdgcn_wmma_f32_16x16x32_bf16(
                    false, afr, false, bfr[t], (short)0, acc[s][t],
                    false, false);
        }
    }

    // Epilogue: C/D layout — VGPR r holds M = r + 8*hi, N = lh.
    float* ob = out + (size_t)batch * (V * D);
#pragma unroll
    for (int s = 0; s < 4; ++s) {
#pragma unroll
        for (int r = 0; r < 8; ++r) {
            const int j  = m + s * 16 + hi * 8 + r;
            const float rv = rinv[j];
#pragma unroll
            for (int t = 0; t < 4; ++t) {
                const int col = t * 16 + lh;
                float o = acc[s][t][r] * rv + bias[col];
                ob[(size_t)j * D + col] = fmaxf(o, 0.f);
            }
        }
    }
}

// =========================================================================
extern "C" void kernel_launch(void* const* d_in, const int* in_sizes, int n_in,
                              void* d_out, int out_size, void* d_ws, size_t ws_size,
                              hipStream_t stream) {
    const float* adj  = (const float*)d_in[0];   // [2048,2048]
    const float* x    = (const float*)d_in[1];   // [8,8,2048,64]
    const float* wgt  = (const float*)d_in[2];   // [64,64]
    const float* bias = (const float*)d_in[3];   // [64]
    float* out = (float*)d_out;                  // [8,8,2048,64] fp32

    char* wsb = (char*)d_ws;
    float*          rinv = (float*)wsb;                         // 8 KB
    unsigned short* adjT = (unsigned short*)(wsb + ADJT_OFF);   // 8 MB bf16
    unsigned short* tT   = (unsigned short*)(wsb + TT_OFF);     // 16 MB bf16

    colsum_rinv_kernel   <<<dim3(64),      dim3(32, 8), 0, stream>>>(adj, rinv);
    transpose_scale_kernel<<<dim3(64, 64), dim3(32, 8), 0, stream>>>(adj, rinv, adjT);
    project_kernel       <<<dim3(64, 64),  dim3(256),   0, stream>>>(x, wgt, tT);
    gemm_agg_kernel      <<<dim3(256),     dim3(256),   0, stream>>>(adjT, tT, rinv, bias, out);
}
